// MHA_85822036508826
// MI455X (gfx1250) — compile-verified, matrix operand order checked by reading, not run on hardware
//
#include <hip/hip_runtime.h>
#include <math.h>

typedef __attribute__((ext_vector_type(16))) __bf16 v16bf;
typedef __attribute__((ext_vector_type(8)))  float  v8f;

#define S_LEN 4096
#define T_LEN 4096
#define NH    12
#define DH    64
#define HD    (NH * DH)   // 768
#define KT    64          // keys per iteration
#define QW    16          // query rows per wave
#define WAVES 8
#define QB    (QW * WAVES) // 128 query rows per block

union FragBF { v16bf v; unsigned u[8]; uint4 q[2]; };
union FragF  { v8f   v; float   f[8]; };

__device__ inline unsigned pack_bf2(float x, float y) {
    __bf16 a = (__bf16)x, b = (__bf16)y;
    unsigned short ua = __builtin_bit_cast(unsigned short, a);
    unsigned short ub = __builtin_bit_cast(unsigned short, b);
    return (unsigned)ua | ((unsigned)ub << 16);
}

__global__ __launch_bounds__(256)
void fa_fwd_bf16wmma(const float* __restrict__ q, const float* __restrict__ k,
                     const float* __restrict__ v, float* __restrict__ out)
{
    // Double-buffered tiles: one barrier per iteration.
    __shared__ __align__(16) __bf16 Kt[2][KT * DH];   // [buf][key][d]  64x64
    __shared__ __align__(16) __bf16 Vt[2][DH * KT];   // [buf][d][key]  64x64

    const int tid  = threadIdx.x;
    const int lane = tid & 31;
    const int wid  = tid >> 5;
    const int l16  = lane & 15;
    const int half = lane >> 4;
    const int h    = blockIdx.y;
    const int qbase = blockIdx.x * QB + wid * QW;
    const int kb   = half * 8;   // K-base of 16-bit fragment layout for this half

    // ---- Q^T B-fragments, pre-scaled by (1/sqrt(D)) * log2(e)  ----
    // Scores are then in log2-domain; exp2 replaces exp (softmax invariant).
    const float qscale = 0.125f * 1.4426950408889634f;
    FragBF bQ[2];
    {
        const float* qrow = q + (size_t)(qbase + l16) * HD + h * DH;
#pragma unroll
        for (int c = 0; c < 2; ++c) {
            const float* p0 = qrow + c * 32 + kb;
            const float* p1 = qrow + c * 32 + kb + 16;
            float4 a0 = ((const float4*)p0)[0], a1 = ((const float4*)p0)[1];
            float4 b0 = ((const float4*)p1)[0], b1 = ((const float4*)p1)[1];
            bQ[c].u[0] = pack_bf2(a0.x * qscale, a0.y * qscale);
            bQ[c].u[1] = pack_bf2(a0.z * qscale, a0.w * qscale);
            bQ[c].u[2] = pack_bf2(a1.x * qscale, a1.y * qscale);
            bQ[c].u[3] = pack_bf2(a1.z * qscale, a1.w * qscale);
            bQ[c].u[4] = pack_bf2(b0.x * qscale, b0.y * qscale);
            bQ[c].u[5] = pack_bf2(b0.z * qscale, b0.w * qscale);
            bQ[c].u[6] = pack_bf2(b1.x * qscale, b1.y * qscale);
            bQ[c].u[7] = pack_bf2(b1.z * qscale, b1.w * qscale);
        }
    }

    // ---- Online-softmax state: lane-local (one query = l16 per lane) ----
    float m_run = -INFINITY, l_run = 0.0f;
    FragF o[4];  // O^T tiles: lane = query, rows = d
#pragma unroll
    for (int n = 0; n < 4; ++n)
#pragma unroll
        for (int r = 0; r < 8; ++r) o[n].f[r] = 0.0f;

    // ---- Staging mapping: thread -> (key row, 16-wide d block) ----
    const int key = tid >> 2;        // 0..63
    const int db  = (tid & 3) * 16;  // 0,16,32,48
    float4 rk[4], rv[4];

    // prefetch tile 0
    {
        const float* kp = k + (size_t)key * HD + h * DH + db;
        const float* vp = v + (size_t)key * HD + h * DH + db;
#pragma unroll
        for (int j = 0; j < 4; ++j) { rk[j] = ((const float4*)kp)[j]; rv[j] = ((const float4*)vp)[j]; }
    }
    // stage tile 0 into buf 0
    {
        unsigned* kd = (unsigned*)&Kt[0][key * DH + db];
#pragma unroll
        for (int j = 0; j < 4; ++j) {
            kd[2 * j]     = pack_bf2(rk[j].x, rk[j].y);
            kd[2 * j + 1] = pack_bf2(rk[j].z, rk[j].w);
        }
        float vv[16] = {rv[0].x, rv[0].y, rv[0].z, rv[0].w, rv[1].x, rv[1].y, rv[1].z, rv[1].w,
                        rv[2].x, rv[2].y, rv[2].z, rv[2].w, rv[3].x, rv[3].y, rv[3].z, rv[3].w};
#pragma unroll
        for (int j = 0; j < 16; ++j) Vt[0][(db + j) * KT + key] = (__bf16)vv[j];
    }
    __syncthreads();

    int p = 0;
    for (int t0 = 0; t0 < T_LEN; t0 += KT) {
        const bool more = (t0 + KT) < T_LEN;

        // ---- Prefetch next tile into registers (hides under compute) ----
        if (more) {
            const float* kp = k + (size_t)(t0 + KT + key) * HD + h * DH + db;
            const float* vp = v + (size_t)(t0 + KT + key) * HD + h * DH + db;
#pragma unroll
            for (int j = 0; j < 4; ++j) { rk[j] = ((const float4*)kp)[j]; rv[j] = ((const float4*)vp)[j]; }
        }

        // ---- Scores (transposed): S^T = K . Q^T, 4 tiles of 16 keys ----
        FragF s[4];
#pragma unroll
        for (int kt = 0; kt < 4; ++kt) {
#pragma unroll
            for (int i = 0; i < 8; ++i) s[kt].f[i] = 0.0f;
#pragma unroll
            for (int c = 0; c < 2; ++c) {
                FragBF aK;  // A: 16 keys x 32 d
                const __bf16* kr = &Kt[p][(kt * 16 + l16) * DH + c * 32];
                aK.q[0] = *(const uint4*)(kr + kb);
                aK.q[1] = *(const uint4*)(kr + kb + 16);
                s[kt].v = __builtin_amdgcn_wmma_f32_16x16x32_bf16(
                    false, aK.v, false, bQ[c].v, (short)0, s[kt].v, false, false);
            }
        }

        // ---- Online softmax: local max tree + ONE cross-half shuffle ----
        float mloc = s[0].f[0];
#pragma unroll
        for (int kt = 0; kt < 4; ++kt)
#pragma unroll
            for (int i = 0; i < 8; ++i) mloc = fmaxf(mloc, s[kt].f[i]);
        mloc = fmaxf(mloc, __shfl_xor(mloc, 16, 32));

        float mnew = fmaxf(m_run, mloc);
        float corr = __builtin_amdgcn_exp2f(m_run - mnew);
        m_run = mnew;

        float e[4][8], rs = 0.0f;
#pragma unroll
        for (int kt = 0; kt < 4; ++kt)
#pragma unroll
            for (int r = 0; r < 8; ++r) {
                e[kt][r] = __builtin_amdgcn_exp2f(s[kt].f[r] - mnew);
                rs += e[kt][r];
            }
        rs += __shfl_xor(rs, 16, 32);
        l_run = l_run * corr + rs;

#pragma unroll
        for (int n = 0; n < 4; ++n)
#pragma unroll
            for (int r = 0; r < 8; ++r) o[n].f[r] *= corr;

        // ---- P^T B-fragments: pure register packing ----
        FragBF bP[2];
#pragma unroll
        for (int g = 0; g < 2; ++g)
#pragma unroll
            for (int j = 0; j < 4; ++j) {
                bP[g].u[j]     = pack_bf2(e[2 * g][2 * j],     e[2 * g][2 * j + 1]);
                bP[g].u[4 + j] = pack_bf2(e[2 * g + 1][2 * j], e[2 * g + 1][2 * j + 1]);
            }

        // ---- O^T += V^T . P^T  (4 d-tiles x 2 key-chunks) ----
#pragma unroll
        for (int n = 0; n < 4; ++n)
#pragma unroll
            for (int g = 0; g < 2; ++g) {
                FragBF aV;  // A: 16 d x 32 keys
                const __bf16* vr = &Vt[p][(n * 16 + l16) * KT + g * 32];
                aV.q[0] = *(const uint4*)(vr + kb);
                aV.q[1] = *(const uint4*)(vr + kb + 16);
                o[n].v = __builtin_amdgcn_wmma_f32_16x16x32_bf16(
                    false, aV.v, false, bP[g].v, (short)0, o[n].v, false, false);
            }

        // ---- Stage prefetched tile into the other buffer; single barrier ----
        if (more) {
            int np = p ^ 1;
            unsigned* kd = (unsigned*)&Kt[np][key * DH + db];
#pragma unroll
            for (int j = 0; j < 4; ++j) {
                kd[2 * j]     = pack_bf2(rk[j].x, rk[j].y);
                kd[2 * j + 1] = pack_bf2(rk[j].z, rk[j].w);
            }
            float vv[16] = {rv[0].x, rv[0].y, rv[0].z, rv[0].w, rv[1].x, rv[1].y, rv[1].z, rv[1].w,
                            rv[2].x, rv[2].y, rv[2].z, rv[2].w, rv[3].x, rv[3].y, rv[3].z, rv[3].w};
#pragma unroll
            for (int j = 0; j < 16; ++j) Vt[np][(db + j) * KT + key] = (__bf16)vv[j];
            __syncthreads();
            p = np;
        }
    }

    // ---- Epilogue: lane-local normalize, contiguous float4 stores ----
    {
        float inv = 1.0f / l_run;
        float* orow = out + (size_t)(qbase + l16) * HD + h * DH + 8 * half;
#pragma unroll
        for (int n = 0; n < 4; ++n) {
            float4 x0 = {o[n].f[0] * inv, o[n].f[1] * inv, o[n].f[2] * inv, o[n].f[3] * inv};
            float4 x1 = {o[n].f[4] * inv, o[n].f[5] * inv, o[n].f[6] * inv, o[n].f[7] * inv};
            ((float4*)(orow + n * 16))[0] = x0;
            ((float4*)(orow + n * 16 + 4))[0] = x1;
        }
    }
}

extern "C" void kernel_launch(void* const* d_in, const int* in_sizes, int n_in,
                              void* d_out, int out_size, void* d_ws, size_t ws_size,
                              hipStream_t stream) {
    (void)in_sizes; (void)n_in; (void)out_size; (void)d_ws; (void)ws_size;
    const float* q = (const float*)d_in[0];
    const float* k = (const float*)d_in[1];
    const float* v = (const float*)d_in[2];
    float* out = (float*)d_out;
    dim3 grid(S_LEN / QB, NH);
    fa_fwd_bf16wmma<<<grid, 256, 0, stream>>>(q, k, v, out);
}
